// VitCorrectModule_72172630442484
// MI455X (gfx1250) — matrix operand order repaired and checked by reading, not compiled, not run on hardware
//
#include <hip/hip_runtime.h>
#include <hip/hip_bf16.h>

#define BSZ   2
#define LQ    4096
#define CDIM  768
#define NHEAD 6
#define HDIM  128
#define HW    4096
#define MROWS 8192      // B*Lq == B*HW
#define NPAD  128       // padded off+attn GEMM width (48 off + 24 attn + pad)

typedef __attribute__((ext_vector_type(16))) __bf16 v16bf;
typedef __attribute__((ext_vector_type(8)))  float  v8f;

__device__ __forceinline__ unsigned short f2bf(float f) {
  unsigned u = __float_as_uint(f);
  u += 0x7FFFu + ((u >> 16) & 1u);          // round-to-nearest-even
  return (unsigned short)(u >> 16);
}
__device__ __forceinline__ float bf2f(unsigned short s) {
  return __uint_as_float(((unsigned)s) << 16);
}

// ---------------------------------------------------------------------------
// LayerNorm(x)->q_bf16 and LayerNorm(feat)->f_bf16, one row per block.
// ---------------------------------------------------------------------------
__global__ __launch_bounds__(256) void ln_to_bf16(
    const float* __restrict__ x, const float* __restrict__ feat,
    const float* __restrict__ qw, const float* __restrict__ qb,
    const float* __restrict__ fw, const float* __restrict__ fb,
    unsigned short* __restrict__ qbf, unsigned short* __restrict__ fbf)
{
  __shared__ float red[256];
  const int row = blockIdx.x;                 // 0..16383
  const bool isF = row >= MROWS;
  const int r = isF ? row - MROWS : row;
  const float* src = (isF ? feat : x) + (size_t)r * CDIM;
  const float* w   = isF ? fw : qw;
  const float* bb  = isF ? fb : qb;
  unsigned short* dst = (isF ? fbf : qbf) + (size_t)r * CDIM;

  const int t = threadIdx.x;
  float v0 = src[t], v1 = src[t + 256], v2 = src[t + 512];
  red[t] = v0 + v1 + v2;
  __syncthreads();
  for (int o = 128; o > 0; o >>= 1) { if (t < o) red[t] += red[t + o]; __syncthreads(); }
  const float mean = red[0] * (1.f / CDIM);
  __syncthreads();
  float d0 = v0 - mean, d1 = v1 - mean, d2 = v2 - mean;
  red[t] = d0 * d0 + d1 * d1 + d2 * d2;
  __syncthreads();
  for (int o = 128; o > 0; o >>= 1) { if (t < o) red[t] += red[t + o]; __syncthreads(); }
  const float rs = rsqrtf(red[0] * (1.f / CDIM) + 1e-6f);
  dst[t]       = f2bf(d0 * rs * w[t]       + bb[t]);
  dst[t + 256] = f2bf(d1 * rs * w[t + 256] + bb[t + 256]);
  dst[t + 512] = f2bf(d2 * rs * w[t + 512] + bb[t + 512]);
}

// ---------------------------------------------------------------------------
// Transpose fp32 [K=768][ncols] weight -> bf16 [ncols][768].
// ---------------------------------------------------------------------------
__global__ __launch_bounds__(256) void transpose_to_bf16(
    const float* __restrict__ w, unsigned short* __restrict__ wt, int ncols)
{
  int idx = blockIdx.x * 256 + threadIdx.x;
  if (idx >= ncols * CDIM) return;
  int n = idx / CDIM, k = idx - n * CDIM;
  wt[idx] = f2bf(w[(size_t)k * ncols + n]);
}

// ---------------------------------------------------------------------------
// Pack w_off[768][48] + w_attn[768][24] into zero-padded bf16 [128][768]
// (transposed), and b_off/b_attn into padded fp32 bias[128].
// ---------------------------------------------------------------------------
__global__ __launch_bounds__(256) void pack_offattn(
    const float* __restrict__ woff, const float* __restrict__ watt,
    const float* __restrict__ boff, const float* __restrict__ batt,
    unsigned short* __restrict__ wt, float* __restrict__ biasPad)
{
  int idx = blockIdx.x * 256 + threadIdx.x;
  if (idx < NPAD * CDIM) {
    int n = idx / CDIM, k = idx - n * CDIM;
    float v = 0.f;
    if (n < 48)      v = woff[(size_t)k * 48 + n];
    else if (n < 72) v = watt[(size_t)k * 24 + (n - 48)];
    wt[idx] = f2bf(v);
  }
  if (idx < NPAD) {
    float bv = 0.f;
    if (idx < 48)      bv = boff[idx];
    else if (idx < 72) bv = batt[idx - 48];
    biasPad[idx] = bv;
  }
}

// ---------------------------------------------------------------------------
// bf16 WMMA GEMM: C[M x N] = A[M x K] * Bt[N x K]^T (+bias, epilogue modes).
// Block = 256 thr (8 waves), tile 128x128, wave tile 64x32 (4x2 wmma accs).
// Global -> LDS staging uses GLOBAL_LOAD_ASYNC_TO_LDS_B128 (ASYNCcnt) with a
// double-buffered LDS tile: one barrier per K-step, loads overlap WMMAs.
// mode 0: Cf[r*N+c]  = acc + bias[c]
// mode 1: Cf[r*N+c]  = xres[r*N+c] + gamma[c]*(acc + bias[c])
// mode 2: value scatter: Cbf[((b*NH+head)*HW+hw)*HD+d] = bf16(acc+bias[c])
// ---------------------------------------------------------------------------
__global__ __launch_bounds__(256) void gemm_bf16_wmma(
    const unsigned short* __restrict__ A,
    const unsigned short* __restrict__ Bt,
    const float* __restrict__ bias,
    float* __restrict__ Cf,
    unsigned short* __restrict__ Cbf,
    const float* __restrict__ xres,
    const float* __restrict__ gamma,
    int N, int K, int mode)
{
  // [sel][ A(8KB) | B(8KB) ]  -> 32KB total
  __shared__ __align__(16) unsigned short smem[2 * 2 * 128 * 32];

  const int t    = threadIdx.x;
  const int lane = t & 31;
  const int wid  = t >> 5;
  const int wm   = (wid & 1) * 64;      // wave M offset in tile
  const int wn   = (wid >> 1) * 32;     // wave N offset in tile
  const int rowBase = blockIdx.y * 128;
  const int colBase = blockIdx.x * 128;

  v8f acc[4][2];
#pragma unroll
  for (int i = 0; i < 4; ++i)
#pragma unroll
    for (int j = 0; j < 2; ++j)
#pragma unroll
      for (int e = 0; e < 8; ++e) acc[i][j][e] = 0.f;

  // Loader: thread t stages 16 bf16 of A and 16 of Bt per 32-wide K step
  // (two async b128 per matrix per wave).
  const int ldr = t >> 1;               // 0..127 (row of A tile / row of Bt tile)
  const int ldc = (t & 1) * 16;         // 0 or 16
  const unsigned short* Ag = A  + (size_t)(rowBase + ldr) * K + ldc;
  const unsigned short* Bg = Bt + (size_t)(colBase + ldr) * K + ldc;

  const unsigned ldsBase = (unsigned)(uintptr_t)(void*)smem;  // LDS byte offset
  const unsigned stOff   = (unsigned)((ldr * 32 + ldc) * 2);

  auto issue = [&](int sel, int k0) {
    const unsigned la = ldsBase + (unsigned)sel * 16384u + stOff;
    const unsigned lb = la + 8192u;
    const unsigned long long ga = (unsigned long long)(uintptr_t)(Ag + k0);
    const unsigned long long gb = (unsigned long long)(uintptr_t)(Bg + k0);
    asm volatile("global_load_async_to_lds_b128 %0, %1, off"
                 :: "v"(la),        "v"(ga)      : "memory");
    asm volatile("global_load_async_to_lds_b128 %0, %1, off"
                 :: "v"(la + 16u),  "v"(ga + 16) : "memory");
    asm volatile("global_load_async_to_lds_b128 %0, %1, off"
                 :: "v"(lb),        "v"(gb)      : "memory");
    asm volatile("global_load_async_to_lds_b128 %0, %1, off"
                 :: "v"(lb + 16u),  "v"(gb + 16) : "memory");
  };

  // Per-lane fragment byte offsets (row stride in LDS = 32 bf16 = 64B).
  // A frag (16x32): lanes<16 hold K0..7|K16..23, lanes>=16 K8..15|K24..31.
  // B frag (32x16): lanes<16 hold K0..7|K16..23, lanes>=16 K8..15|K24..31.
  const char* smemB = (const char*)smem;
  const int aoff = (wm + (lane & 15)) * 64 + (lane >> 4) * 16;
  const int boff = 8192 + (wn + (lane & 15)) * 64 + (lane >> 4) * 16;

  const int NK = K >> 5;
  issue(0, 0);
  for (int it = 0; it < NK; ++it) {
    const int sel = it & 1;
    asm volatile("s_wait_asynccnt 0" ::: "memory");  // my tile[sel] loads done
    __syncthreads();                                  // everyone's loads done,
                                                      // everyone's reads of tile[sel^1] done
    if (it + 1 < NK) issue(sel ^ 1, (it + 1) << 5);   // overlap with compute

    const char* sb = smemB + sel * 16384;
    union { v16bf v; uint4 q[2]; } af[4], bfm[2];
#pragma unroll
    for (int i = 0; i < 4; ++i) {
      af[i].q[0] = *(const uint4*)(sb + aoff + i * 16 * 64);
      af[i].q[1] = *(const uint4*)(sb + aoff + i * 16 * 64 + 32);
    }
#pragma unroll
    for (int j = 0; j < 2; ++j) {
      bfm[j].q[0] = *(const uint4*)(sb + boff + j * 16 * 64);
      bfm[j].q[1] = *(const uint4*)(sb + boff + j * 16 * 64 + 32);
    }
#pragma unroll
    for (int i = 0; i < 4; ++i)
#pragma unroll
      for (int j = 0; j < 2; ++j)
        acc[i][j] = __builtin_amdgcn_wmma_f32_16x16x32_bf16(
            false, af[i].v, false, bfm[j].v, (short)0, acc[i][j], false, false);
  }

  // Epilogue: element v of acc -> row = v + 8*(lane>>4), col = lane&15.
#pragma unroll
  for (int i = 0; i < 4; ++i) {
#pragma unroll
    for (int j = 0; j < 2; ++j) {
      const int r0 = rowBase + wm + i * 16 + (lane >> 4) * 8;
      const int c  = colBase + wn + j * 16 + (lane & 15);
      const float bv = bias[c];
#pragma unroll
      for (int v = 0; v < 8; ++v) {
        const int r = r0 + v;
        const float val = acc[i][j][v] + bv;
        if (mode == 0) {
          Cf[(size_t)r * N + c] = val;
        } else if (mode == 1) {
          const size_t idx = (size_t)r * N + c;
          Cf[idx] = xres[idx] + gamma[c] * val;
        } else {
          const int b = r >> 12, hw = r & 4095;
          const int head = c >> 7, d = c & 127;
          Cbf[(((size_t)(b * NHEAD + head)) * HW + hw) * HDIM + d] = f2bf(val);
        }
      }
    }
  }
}

// ---------------------------------------------------------------------------
// Per (b,h,q): softmax over the 4 attention logits, compute pixel coords.
// samp layout: [(b*NH+h)*LQ + q][p][{px,py,w}]  (12 floats)
// ---------------------------------------------------------------------------
__global__ __launch_bounds__(256) void softmax_loc(
    const float* __restrict__ OA, const float* __restrict__ refp,
    float* __restrict__ samp)
{
  const int tt = blockIdx.x * 256 + threadIdx.x;
  if (tt >= BSZ * NHEAD * LQ) return;
  const int q  = tt & 4095;
  const int bh = tt >> 12;
  const int h  = bh % NHEAD;
  const int b  = bh / NHEAD;
  const int r  = b * LQ + q;
  const float* oa = OA + (size_t)r * NPAD;
  const float rx = refp[(size_t)r * 2 + 0] * 64.f - 0.5f;
  const float ry = refp[(size_t)r * 2 + 1] * 64.f - 0.5f;
  float a[4]; float m = -1e30f;
#pragma unroll
  for (int p = 0; p < 4; ++p) { a[p] = oa[48 + h * 4 + p]; m = fmaxf(m, a[p]); }
  float s = 0.f;
#pragma unroll
  for (int p = 0; p < 4; ++p) { a[p] = __expf(a[p] - m); s += a[p]; }
  const float inv = 1.f / s;
  float* o = samp + (size_t)tt * 12;
#pragma unroll
  for (int p = 0; p < 4; ++p) {
    o[p * 3 + 0] = rx + oa[h * 8 + p * 2 + 0];
    o[p * 3 + 1] = ry + oa[h * 8 + p * 2 + 1];
    o[p * 3 + 2] = a[p] * inv;
  }
}

// ---------------------------------------------------------------------------
// One wave per (b,h,q): bilinear-gather 4 points x 4 corners from the
// head-contiguous bf16 value tensor, each lane owns 4 of 128 head dims.
// Writes attn as bf16 rows [B*Lq][768] for the final WMMA GEMM.
// ---------------------------------------------------------------------------
__global__ __launch_bounds__(256) void deform_gather(
    const unsigned short* __restrict__ vbf,   // [B][NH][HW][HD] bf16
    const float* __restrict__ samp,
    unsigned short* __restrict__ attnbf)      // [B*Lq][768] bf16
{
  const int gw   = (int)((blockIdx.x * 256u + threadIdx.x) >> 5);
  const int lane = threadIdx.x & 31;
  const int q  = gw & 4095;
  const int bh = gw >> 12;                    // b*NH + h
  const float* sp = samp + (size_t)gw * 12;
  const unsigned short* vb = vbf + (size_t)bh * HW * HDIM + lane * 4;

  float a0 = 0.f, a1 = 0.f, a2 = 0.f, a3 = 0.f;
#pragma unroll
  for (int p = 0; p < 4; ++p) {
    const float px = sp[p * 3 + 0], py = sp[p * 3 + 1], w = sp[p * 3 + 2];
    const float fx0 = floorf(px), fy0 = floorf(py);
    const float fx = px - fx0, fy = py - fy0;
    const int x0 = (int)fx0, y0 = (int)fy0;
#pragma unroll
    for (int c = 0; c < 4; ++c) {
      const int dy = c >> 1, dx = c & 1;
      const int yi = y0 + dy, xi = x0 + dx;
      const float cw = w * (dy ? fy : 1.f - fy) * (dx ? fx : 1.f - fx);
      if (yi >= 0 && yi < 64 && xi >= 0 && xi < 64) {   // wave-uniform branch
        const unsigned short* pv = vb + (size_t)(yi * 64 + xi) * HDIM;
        const uint2 raw = *(const uint2*)pv;
        a0 += cw * bf2f((unsigned short)(raw.x & 0xFFFFu));
        a1 += cw * bf2f((unsigned short)(raw.x >> 16));
        a2 += cw * bf2f((unsigned short)(raw.y & 0xFFFFu));
        a3 += cw * bf2f((unsigned short)(raw.y >> 16));
      }
    }
  }
  const int b = bh / NHEAD, h = bh % NHEAD;
  unsigned short* dst = attnbf + ((size_t)(b * LQ + q)) * CDIM + h * HDIM + lane * 4;
  const unsigned r0 = (unsigned)f2bf(a0) | ((unsigned)f2bf(a1) << 16);
  const unsigned r1 = (unsigned)f2bf(a2) | ((unsigned)f2bf(a3) << 16);
  *(uint2*)dst = make_uint2(r0, r1);
}

// ---------------------------------------------------------------------------
extern "C" void kernel_launch(void* const* d_in, const int* in_sizes, int n_in,
                              void* d_out, int out_size, void* d_ws, size_t ws_size,
                              hipStream_t stream)
{
  (void)in_sizes; (void)n_in; (void)out_size; (void)ws_size;
  const float* x       = (const float*)d_in[0];
  const float* refp    = (const float*)d_in[1];
  const float* feat    = (const float*)d_in[2];
  // d_in[3] spatial_shapes, d_in[4] level_start_index: compile-time constants here
  const float* qn_w    = (const float*)d_in[5];
  const float* qn_b    = (const float*)d_in[6];
  const float* fn_w    = (const float*)d_in[7];
  const float* fn_b    = (const float*)d_in[8];
  const float* gamma   = (const float*)d_in[9];
  const float* w_value = (const float*)d_in[10];
  const float* b_value = (const float*)d_in[11];
  const float* w_off   = (const float*)d_in[12];
  const float* b_off   = (const float*)d_in[13];
  const float* w_attn  = (const float*)d_in[14];
  const float* b_attn  = (const float*)d_in[15];
  const float* w_out   = (const float*)d_in[16];
  const float* b_out   = (const float*)d_in[17];
  float* out = (float*)d_out;

  char* ws = (char*)d_ws;
  size_t o = 0;
  auto alloc = [&](size_t bytes) {
    char* p = ws + o; o += (bytes + 255) & ~(size_t)255; return p;
  };
  unsigned short* qbf    = (unsigned short*)alloc((size_t)MROWS * CDIM * 2);
  unsigned short* fbf    = (unsigned short*)alloc((size_t)MROWS * CDIM * 2);
  unsigned short* wTv    = (unsigned short*)alloc((size_t)CDIM * CDIM * 2);
  unsigned short* wTo    = (unsigned short*)alloc((size_t)CDIM * CDIM * 2);
  unsigned short* wToa   = (unsigned short*)alloc((size_t)NPAD * CDIM * 2);
  float*          biasPad= (float*)         alloc((size_t)NPAD * 4);
  unsigned short* valbf  = (unsigned short*)alloc((size_t)MROWS * CDIM * 2);
  float*          OA     = (float*)         alloc((size_t)MROWS * NPAD * 4);
  float*          samp   = (float*)         alloc((size_t)BSZ * NHEAD * LQ * 12 * 4);
  unsigned short* attnbf = (unsigned short*)alloc((size_t)MROWS * CDIM * 2);

  // 1) LayerNorms -> bf16 activations
  ln_to_bf16<<<2 * MROWS, 256, 0, stream>>>(x, feat, qn_w, qn_b, fn_w, fn_b, qbf, fbf);

  // 2) Weight prep (bf16, transposed / padded)
  transpose_to_bf16<<<(CDIM * CDIM + 255) / 256, 256, 0, stream>>>(w_value, wTv, CDIM);
  transpose_to_bf16<<<(CDIM * CDIM + 255) / 256, 256, 0, stream>>>(w_out,   wTo, CDIM);
  pack_offattn<<<(NPAD * CDIM + 255) / 256, 256, 0, stream>>>(
      w_off, w_attn, b_off, b_attn, wToa, biasPad);

  // 3) value = LN(feat) @ w_value + b_value  -> [B][NH][HW][HD] bf16 (mode 2)
  gemm_bf16_wmma<<<dim3(CDIM / 128, MROWS / 128), 256, 0, stream>>>(
      fbf, wTv, b_value, nullptr, valbf, nullptr, nullptr, CDIM, CDIM, 2);

  // 4) off/attn logits = q @ [w_off|w_attn] + bias -> [MROWS][128] f32 (mode 0)
  gemm_bf16_wmma<<<dim3(NPAD / 128, MROWS / 128), 256, 0, stream>>>(
      qbf, wToa, biasPad, OA, nullptr, nullptr, nullptr, NPAD, CDIM, 0);

  // 5) softmax + sampling coords
  softmax_loc<<<(BSZ * NHEAD * LQ + 255) / 256, 256, 0, stream>>>(OA, refp, samp);

  // 6) deformable bilinear gather -> attn bf16
  deform_gather<<<(BSZ * NHEAD * LQ) / 8, 256, 0, stream>>>(valbf, samp, attnbf);

  // 7) out = x + gamma * (attn @ w_out + b_out)   (mode 1, writes d_out f32)
  gemm_bf16_wmma<<<dim3(CDIM / 128, MROWS / 128), 256, 0, stream>>>(
      attnbf, wTo, b_out, out, nullptr, x, gamma, CDIM, CDIM, 1);
}
